// DecoderBlock_33998961115535
// MI455X (gfx1250) — matrix operand order repaired and checked by reading, not compiled, not run on hardware
//
#include <hip/hip_runtime.h>
#include <cstdint>
#include <cstddef>

typedef __attribute__((ext_vector_type(8)))  float   v8f;
typedef __attribute__((ext_vector_type(8)))  __bf16  v8bf;
typedef __attribute__((ext_vector_type(16))) __bf16  v16bf;

#define D_MODEL 512
#define NEGINF  (-1e10f)

// Native cast -> compiler selects v_cvt_pk_bf16_f32 (RNE) on gfx1250.
static __device__ __forceinline__ __bf16 tobf(float f) { return (__bf16)f; }

static __device__ __forceinline__ v8f zero8() {
  v8f z;
#pragma unroll
  for (int i = 0; i < 8; ++i) z[i] = 0.0f;
  return z;
}

// CDNA5 16-bit A-matrix fragment (16x32): lane holds row m=lane%16.
// element e: K = e + 8*(e>=8) + 8*half  -> two contiguous 8-elem runs.
static __device__ __forceinline__ v16bf lds_a_frag(const __bf16* p, int half) {
  v8bf lo = *(const v8bf*)(p + 8 * half);
  v8bf hi = *(const v8bf*)(p + 16 + 8 * half);
  return __builtin_shufflevector(lo, hi, 0, 1, 2, 3, 4, 5, 6, 7,
                                 8, 9, 10, 11, 12, 13, 14, 15);
}
// 16-bit B-matrix fragment (32x16): lane holds col n=lane%16,
// element e: K = e + 8*(e>=8) + 8*half -> two contiguous 8-elem runs.
static __device__ __forceinline__ v16bf lds_b_frag(const __bf16* p, int half) {
  v8bf lo = *(const v8bf*)(p + 8 * half);
  v8bf hi = *(const v8bf*)(p + 16 + 8 * half);
  return __builtin_shufflevector(lo, hi, 0, 1, 2, 3, 4, 5, 6, 7,
                                 8, 9, 10, 11, 12, 13, 14, 15);
}

static __device__ __forceinline__ v8f wmma_bf16(v16bf a, v16bf b, v8f c) {
  return __builtin_amdgcn_wmma_f32_16x16x32_bf16(false, a, false, b,
                                                 (short)0, c, false, false);
}

// -------------------------------------------------------------------------
// C[M,N] = A[M,K] @ W[N,K]^T + bias (torch Linear), optional ReLU.
// 256 threads = 8 waves; 128x128 tile (32 flop/byte from global);
// K staged in 32-chunks as bf16 in LDS.
// -------------------------------------------------------------------------
__global__ __launch_bounds__(256) void gemm_bias_kernel(
    const float* __restrict__ A, const float* __restrict__ W,
    const float* __restrict__ bias, float* __restrict__ C,
    int M, int N, int K, int relu) {
  __shared__ __align__(64) __bf16 As[128 * 32];
  __shared__ __align__(64) __bf16 Ws[128 * 32];
  const int tid = threadIdx.x;
  const int lane = tid & 31, wv = tid >> 5;
  const int half = lane >> 4, n = lane & 15;
  const int nbase = blockIdx.x * 128, mbase = blockIdx.y * 128;

  v8f acc[8];
#pragma unroll
  for (int t = 0; t < 8; ++t) acc[t] = zero8();

  const int nk = K >> 5;
  for (int kc = 0; kc < nk; ++kc) {
#pragma unroll
    for (int i = 0; i < 4; ++i) {
      int idx = tid + i * 256;
      int r = idx >> 3, c4 = idx & 7;
      float4 av = *(const float4*)(A + (size_t)(mbase + r) * K + kc * 32 + c4 * 4);
      float4 wf = *(const float4*)(W + (size_t)(nbase + r) * K + kc * 32 + c4 * 4);
      __bf16* ap = As + r * 32 + c4 * 4;
      ap[0] = tobf(av.x); ap[1] = tobf(av.y); ap[2] = tobf(av.z); ap[3] = tobf(av.w);
      __bf16* wp = Ws + r * 32 + c4 * 4;
      wp[0] = tobf(wf.x); wp[1] = tobf(wf.y); wp[2] = tobf(wf.z); wp[3] = tobf(wf.w);
    }
    if (kc + 1 < nk) {  // gfx1250 global_prefetch_b8 hint for next K chunk
      __builtin_prefetch(A + (size_t)(mbase + (tid >> 1)) * K + (kc + 1) * 32, 0, 0);
      __builtin_prefetch(W + (size_t)(nbase + (tid >> 1)) * K + (kc + 1) * 32, 0, 0);
    }
    __syncthreads();
    v16bf a = lds_a_frag(As + (wv * 16 + n) * 32, half);
#pragma unroll
    for (int t = 0; t < 8; ++t) {
      v16bf b = lds_b_frag(Ws + (t * 16 + n) * 32, half);
      acc[t] = wmma_bf16(a, b, acc[t]);
    }
    __syncthreads();
  }
  // C tile layout: vgpr r -> row r + 8*half, lane%16 -> col
#pragma unroll
  for (int t = 0; t < 8; ++t) {
    int col = nbase + t * 16 + n;
    float bv = bias[col];
#pragma unroll
    for (int r = 0; r < 8; ++r) {
      int row = mbase + wv * 16 + r + 8 * half;
      float val = acc[t][r] + bv;
      if (relu) val = fmaxf(val, 0.0f);
      C[(size_t)row * N + col] = val;
    }
  }
}

// -------------------------------------------------------------------------
// Flash attention, d=512, single "head". 4 waves / block, 64 Q rows / block,
// 16 Q rows / wave, 32-row KV blocks. O accumulated in 32 v8f tiles (f32).
// Reference quirks preserved: mask(-1e10) applied BEFORE scale=1/sqrt(Skv).
// -------------------------------------------------------------------------
__global__ __launch_bounds__(128) void flash_attn_kernel(
    const float* __restrict__ q, const float* __restrict__ k,
    const float* __restrict__ v, float* __restrict__ o,
    int Sq, int Skv, int causal) {
  extern __shared__ __align__(128) char smem_raw[];
  __bf16* Qs = (__bf16*)smem_raw;  // 64 x 512
  __bf16* Ks = Qs + 64 * 512;      // 32 x 512 (row major)
  __bf16* Vt = Ks + 32 * 512;      // 512 x 32 (V transposed)
  __bf16* Ps = Vt + 512 * 32;      // 4 x (16 x 32) probs staging

  const int tid = threadIdx.x;
  const int lane = tid & 31, wv = tid >> 5;
  const int half = lane >> 4, n = lane & 15;
  const int qbase = blockIdx.x * 64;
  const int batch = blockIdx.y;
  const float scale = rsqrtf((float)Skv);

  const float4* qg = (const float4*)(q + (size_t)batch * Sq * D_MODEL);
  const float4* kg = (const float4*)(k + (size_t)batch * Skv * D_MODEL);
  const float4* vg = (const float4*)(v + (size_t)batch * Skv * D_MODEL);
  float* og = o + (size_t)batch * Sq * D_MODEL;

  // stage Q tile (64 x 512) as bf16
#pragma unroll 4
  for (int i = 0; i < 64; ++i) {
    int idx = tid + i * 128;
    int r = idx >> 7, c = idx & 127;
    float4 f = qg[(size_t)(qbase + r) * 128 + c];
    __bf16* p = Qs + r * 512 + c * 4;
    p[0] = tobf(f.x); p[1] = tobf(f.y); p[2] = tobf(f.z); p[3] = tobf(f.w);
  }

  float mrun[8], lrun[8], alpha[8];
#pragma unroll
  for (int r = 0; r < 8; ++r) { mrun[r] = -1e30f; lrun[r] = 0.0f; }
  v8f acc[32];
#pragma unroll
  for (int t = 0; t < 32; ++t) acc[t] = zero8();

  __bf16* Psw = Ps + wv * (16 * 32);
  const int mrow = wv * 16 + n;

  const int kv_end = causal ? (qbase + 64) : Skv;
  const int nblk = kv_end >> 5;
  for (int j = 0; j < nblk; ++j) {
    const int jbase = j * 32;
    // stage K row-major, V transposed
#pragma unroll 4
    for (int i = 0; i < 32; ++i) {
      int idx = tid + i * 128;
      int r = idx >> 7, c = idx & 127;
      float4 kf = kg[(size_t)(jbase + r) * 128 + c];
      __bf16* p = Ks + r * 512 + c * 4;
      p[0] = tobf(kf.x); p[1] = tobf(kf.y); p[2] = tobf(kf.z); p[3] = tobf(kf.w);
      float4 vf = vg[(size_t)(jbase + r) * 128 + c];
      Vt[(c * 4 + 0) * 32 + r] = tobf(vf.x);
      Vt[(c * 4 + 1) * 32 + r] = tobf(vf.y);
      Vt[(c * 4 + 2) * 32 + r] = tobf(vf.z);
      Vt[(c * 4 + 3) * 32 + r] = tobf(vf.w);
    }
    __syncthreads();

    // S(16x32) = Q(16x512) Kt : 32 WMMAs
    v8f S0 = zero8(), S1 = zero8();
#pragma unroll
    for (int c = 0; c < 16; ++c) {
      v16bf aq = lds_a_frag(Qs + mrow * 512 + c * 32, half);
      v16bf b0 = lds_b_frag(Ks + n * 512 + c * 32, half);
      v16bf b1 = lds_b_frag(Ks + (16 + n) * 512 + c * 32, half);
      S0 = wmma_bf16(aq, b0, S0);
      S1 = wmma_bf16(aq, b1, S1);
    }

    // online softmax; row reductions across the 16 lanes of each half-wave
#pragma unroll
    for (int r = 0; r < 8; ++r) {
      int rowg = qbase + wv * 16 + r + 8 * half;
      float s0 = S0[r], s1 = S1[r];
      if (causal) {
        if (jbase + n > rowg) s0 = NEGINF;
        if (jbase + 16 + n > rowg) s1 = NEGINF;
      }
      s0 *= scale; s1 *= scale;
      float mx = fmaxf(s0, s1);
      mx = fmaxf(mx, __shfl_xor(mx, 1));
      mx = fmaxf(mx, __shfl_xor(mx, 2));
      mx = fmaxf(mx, __shfl_xor(mx, 4));
      mx = fmaxf(mx, __shfl_xor(mx, 8));
      float mn = fmaxf(mrun[r], mx);
      float al = __expf(mrun[r] - mn);
      float p0 = __expf(s0 - mn), p1 = __expf(s1 - mn);
      float rs = p0 + p1;
      rs += __shfl_xor(rs, 1);
      rs += __shfl_xor(rs, 2);
      rs += __shfl_xor(rs, 4);
      rs += __shfl_xor(rs, 8);
      lrun[r] = lrun[r] * al + rs;
      mrun[r] = mn;
      alpha[r] = al;
      // C-layout -> A-layout conversion via per-wave LDS staging
      Psw[(r + 8 * half) * 32 + n] = tobf(p0);
      Psw[(r + 8 * half) * 32 + 16 + n] = tobf(p1);
    }

    // O = diag(alpha)*O + P(16x32) V(32x512) : 32 WMMAs
    v16bf pf = lds_a_frag(Psw + n * 32, half);
#pragma unroll
    for (int t = 0; t < 32; ++t) {
      v16bf bv = lds_b_frag(Vt + (t * 16 + n) * 32, half);
      v8f a = acc[t];
#pragma unroll
      for (int r = 0; r < 8; ++r) a[r] *= alpha[r];
      acc[t] = wmma_bf16(pf, bv, a);
    }
    __syncthreads();
  }

  float linv[8];
#pragma unroll
  for (int r = 0; r < 8; ++r) linv[r] = 1.0f / lrun[r];
#pragma unroll
  for (int t = 0; t < 32; ++t) {
#pragma unroll
    for (int r = 0; r < 8; ++r) {
      int row = qbase + wv * 16 + r + 8 * half;
      og[(size_t)row * D_MODEL + t * 16 + n] = acc[t][r] * linv[r];
    }
  }
}

// -------------------------------------------------------------------------
// Full-tensor layernorm helpers: h = a + b, accumulate sum/sumsq (h may
// alias b, so no restrict), then finalize + normalize.
// -------------------------------------------------------------------------
__global__ __launch_bounds__(256) void add_reduce_kernel(
    const float4* a, const float4* b, float4* h, float* stats) {
  int i = blockIdx.x * 256 + threadIdx.x;
  float4 x = a[i], y = b[i];
  float4 t;
  t.x = x.x + y.x; t.y = x.y + y.y; t.z = x.z + y.z; t.w = x.w + y.w;
  h[i] = t;
  float s = t.x + t.y + t.z + t.w;
  float ss = t.x * t.x + t.y * t.y + t.z * t.z + t.w * t.w;
#pragma unroll
  for (int off = 16; off > 0; off >>= 1) {
    s += __shfl_xor(s, off);
    ss += __shfl_xor(ss, off);
  }
  __shared__ float red[16];
  int lane = threadIdx.x & 31, wv = threadIdx.x >> 5;
  if (lane == 0) { red[wv] = s; red[8 + wv] = ss; }
  __syncthreads();
  if (wv == 0) {
    float s2 = (lane < 8) ? red[lane] : 0.0f;
    float ss2 = (lane < 8) ? red[8 + lane] : 0.0f;
#pragma unroll
    for (int off = 4; off > 0; off >>= 1) {
      s2 += __shfl_xor(s2, off);
      ss2 += __shfl_xor(ss2, off);
    }
    if (lane == 0) { atomicAdd(stats, s2); atomicAdd(stats + 1, ss2); }
  }
}

__global__ void finalize_stats_kernel(float* stats, float invn) {
  float mu = stats[0] * invn;
  float var = stats[1] * invn - mu * mu;
  stats[2] = mu;
  stats[3] = rsqrtf(var + 1e-5f);
}

__global__ __launch_bounds__(256) void norm_kernel(
    const float4* h, const float* stats, float4* o) {
  int i = blockIdx.x * 256 + threadIdx.x;
  float mu = stats[2], rs = stats[3];
  float4 t = h[i];
  float4 r;
  r.x = (t.x - mu) * rs; r.y = (t.y - mu) * rs;
  r.z = (t.z - mu) * rs; r.w = (t.w - mu) * rs;
  o[i] = r;
}

__global__ void zero_stats_kernel(float* stats) {
  if (threadIdx.x < 16) stats[threadIdx.x] = 0.0f;
}

// -------------------------------------------------------------------------
extern "C" void kernel_launch(void* const* d_in, const int* in_sizes, int n_in,
                              void* d_out, int out_size, void* d_ws, size_t ws_size,
                              hipStream_t stream) {
  const float* y = (const float*)d_in[0];
  const float* x = (const float*)d_in[1];
  const float* wq1 = (const float*)d_in[2];
  const float* bq1 = (const float*)d_in[3];
  const float* wk1 = (const float*)d_in[4];
  const float* bk1 = (const float*)d_in[5];
  const float* wv1 = (const float*)d_in[6];
  const float* bv1 = (const float*)d_in[7];
  const float* wq2 = (const float*)d_in[8];
  const float* bq2 = (const float*)d_in[9];
  const float* wk2 = (const float*)d_in[10];
  const float* bk2 = (const float*)d_in[11];
  const float* wv2 = (const float*)d_in[12];
  const float* bv2 = (const float*)d_in[13];
  const float* wff1 = (const float*)d_in[14];
  const float* bff1 = (const float*)d_in[15];
  const float* wff2 = (const float*)d_in[16];
  const float* bff2 = (const float*)d_in[17];

  const int B = 2, S = 4096, D = 512, H = 2048;
  const int M = B * S;
  const size_t MD = (size_t)M * D;

  float* b0 = (float*)d_ws;          // q / ff2-out
  float* b1 = b0 + MD;               // k
  float* b2 = b1 + MD;               // v
  float* b3 = b2 + MD;               // attn out -> residual sum
  float* b4 = b3 + MD;               // h1
  float* b5 = b4 + MD;               // h2
  float* b6 = b5 + MD;               // ffn hidden (M x H)
  float* stats = b6 + (size_t)M * H; // 3 x 4 floats
  float* out = (float*)d_out;

  dim3 gemm_blk(256);
  dim3 gP(D / 128, M / 128);
  dim3 gF1(H / 128, M / 128);
  dim3 attn_blk(128);
  dim3 gA(S / 64, B);
  const size_t attn_smem =
      (size_t)(64 * 512 + 32 * 512 + 512 * 32 + 4 * 16 * 32) * sizeof(__bf16);
  (void)hipFuncSetAttribute((const void*)flash_attn_kernel,
                            hipFuncAttributeMaxDynamicSharedMemorySize,
                            (int)attn_smem);

  const int n4 = (int)(MD / 4);
  dim3 gR(n4 / 256);
  const float invn = 1.0f / (float)MD;

  zero_stats_kernel<<<1, 32, 0, stream>>>(stats);

  // ---- causal self-attention ----
  gemm_bias_kernel<<<gP, gemm_blk, 0, stream>>>(y, wq1, bq1, b0, M, D, D, 0);
  gemm_bias_kernel<<<gP, gemm_blk, 0, stream>>>(y, wk1, bk1, b1, M, D, D, 0);
  gemm_bias_kernel<<<gP, gemm_blk, 0, stream>>>(y, wv1, bv1, b2, M, D, D, 0);
  flash_attn_kernel<<<gA, attn_blk, attn_smem, stream>>>(b0, b1, b2, b3, S, S, 1);
  add_reduce_kernel<<<gR, 256, 0, stream>>>((const float4*)y, (const float4*)b3,
                                            (float4*)b3, stats);
  finalize_stats_kernel<<<1, 1, 0, stream>>>(stats, invn);
  norm_kernel<<<gR, 256, 0, stream>>>((const float4*)b3, stats, (float4*)b4);

  // ---- cross-attention ----
  gemm_bias_kernel<<<gP, gemm_blk, 0, stream>>>(b4, wq2, bq2, b0, M, D, D, 0);
  gemm_bias_kernel<<<gP, gemm_blk, 0, stream>>>(x, wk2, bk2, b1, M, D, D, 0);
  gemm_bias_kernel<<<gP, gemm_blk, 0, stream>>>(x, wv2, bv2, b2, M, D, D, 0);
  flash_attn_kernel<<<gA, attn_blk, attn_smem, stream>>>(b0, b1, b2, b3, S, S, 0);
  add_reduce_kernel<<<gR, 256, 0, stream>>>((const float4*)b4, (const float4*)b3,
                                            (float4*)b3, stats + 4);
  finalize_stats_kernel<<<1, 1, 0, stream>>>(stats + 4, invn);
  norm_kernel<<<gR, 256, 0, stream>>>((const float4*)b3, stats + 4, (float4*)b5);

  // ---- FFN ----
  gemm_bias_kernel<<<gF1, gemm_blk, 0, stream>>>(b5, wff1, bff1, b6, M, H, D, 1);
  gemm_bias_kernel<<<gP, gemm_blk, 0, stream>>>(b6, wff2, bff2, b0, M, D, H, 0);
  add_reduce_kernel<<<gR, 256, 0, stream>>>((const float4*)b5, (const float4*)b0,
                                            (float4*)b0, stats + 8);
  finalize_stats_kernel<<<1, 1, 0, stream>>>(stats + 8, invn);
  norm_kernel<<<gR, 256, 0, stream>>>((const float4*)b0, stats + 8, (float4*)out);
}
